// GPT_90202903150925
// MI455X (gfx1250) — compile-verified, matrix operand order checked by reading, not run on hardware
//
#include <hip/hip_runtime.h>
#include <math.h>

// ---------------------------------------------------------------------------
// GPT forward for MI455X (gfx1250, wave32, WMMA). bf16 WMMA w/ f32 accumulate.
// Weights pre-converted to bf16 [N,K]; GEMM uses double-buffered LDS with
// split staging (loads issued before compute, stores after) so global-load
// latency hides behind the WMMA stream. 64x128 block, 4 WMMA/wave/K-step.
// ---------------------------------------------------------------------------

typedef __attribute__((ext_vector_type(16))) __bf16 v16bf;
typedef __attribute__((ext_vector_type(8)))  float  v8f;

union BF16x16 { v16bf v; __bf16 e[16]; };
union F32x8   { v8f   v; float  e[8];  };

__device__ inline unsigned pack_bf16(__bf16 a, __bf16 b) {
  union { __bf16 h[2]; unsigned u; } p;
  p.h[0] = a; p.h[1] = b; return p.u;
}

#define B_  2
#define L_  2048
#define D_  512
#define NH_ 8
#define NL_ 4
#define DM_ 2048
#define V_  32000
#define HD_ 64
#define M_  (B_ * L_)     // 4096 rows (batch*seq flattened)
#define S3_ (3 * D_)      // 1536 qkv width

// ----------------------------- embedding -----------------------------------
__global__ __launch_bounds__(128)
void embed_kernel(const int* __restrict__ x,
                  const float* __restrict__ tok,
                  const float* __restrict__ pos,
                  float* __restrict__ h) {
  int row = blockIdx.x;           // 0..M_-1
  int l   = row % L_;
  int t   = x[row];
  const float4* te = (const float4*)(tok + (size_t)t * D_);
  const float4* pe = (const float4*)(pos + (size_t)l * D_);
  float4* hr = (float4*)(h + (size_t)row * D_);
  float4 a = te[threadIdx.x], b = pe[threadIdx.x];
  hr[threadIdx.x] = make_float4(a.x + b.x, a.y + b.y, a.z + b.z, a.w + b.w);
}

// ----------------------------- layernorm -----------------------------------
__global__ __launch_bounds__(128)
void ln_kernel(const float* __restrict__ xin,
               const float* __restrict__ g,
               const float* __restrict__ b,
               float* __restrict__ y) {
  __shared__ float red[128];
  int row = blockIdx.x, tid = threadIdx.x;
  const float4 xv = ((const float4*)(xin + (size_t)row * D_))[tid];
  red[tid] = xv.x + xv.y + xv.z + xv.w;
  __syncthreads();
  for (int off = 64; off > 0; off >>= 1) {
    if (tid < off) red[tid] += red[tid + off];
    __syncthreads();
  }
  float mu = red[0] * (1.0f / D_);
  __syncthreads();
  float4 d = make_float4(xv.x - mu, xv.y - mu, xv.z - mu, xv.w - mu);
  red[tid] = d.x * d.x + d.y * d.y + d.z * d.z + d.w * d.w;
  __syncthreads();
  for (int off = 64; off > 0; off >>= 1) {
    if (tid < off) red[tid] += red[tid + off];
    __syncthreads();
  }
  float rstd = rsqrtf(red[0] * (1.0f / D_) + 1e-5f);
  float4 gv = ((const float4*)g)[tid];
  float4 bv = ((const float4*)b)[tid];
  ((float4*)(y + (size_t)row * D_))[tid] =
      make_float4(d.x * rstd * gv.x + bv.x, d.y * rstd * gv.y + bv.y,
                  d.z * rstd * gv.z + bv.z, d.w * rstd * gv.w + bv.w);
}

// ------------------- weight convert + transpose (fp32 -> bf16) --------------
// Wt[n*K + k] = (bf16) W[k*N + n] ; 64x64 LDS tile transpose.
__global__ __launch_bounds__(256)
void wconv_kernel(const float* __restrict__ W, __bf16* __restrict__ Wt,
                  int K, int N) {
  __shared__ __bf16 sT[64][65];
  int k0 = blockIdx.x * 64, n0 = blockIdx.y * 64;
  int tid = threadIdx.x;
#pragma unroll
  for (int i = 0; i < 16; ++i) {              // read coalesced along n
    int idx = tid + i * 256;                  // 0..4095
    int kk = idx >> 6, n = idx & 63;
    sT[n][kk] = (__bf16)W[(size_t)(k0 + kk) * N + n0 + n];
  }
  __syncthreads();
#pragma unroll
  for (int i = 0; i < 8; ++i) {               // write coalesced along k (pairs)
    int idx = tid + i * 256;                  // 0..2047
    int n = idx >> 5, kp = idx & 31;
    unsigned u = pack_bf16(sT[n][2 * kp], sT[n][2 * kp + 1]);
    *(unsigned*)&Wt[(size_t)(n0 + n) * K + k0 + 2 * kp] = u;
  }
}

// ------------------------------- GEMM --------------------------------------
// C[M,N] = opt_residual + opt_gelu(A[M,K] @ W[K,N] + opt_bias)
// A fp32 row-major; weights as bf16 Wt[N,K]. Block tile 64x128, 8 waves,
// each wave: 1 A-frag reused over 4 B-frags -> 4 WMMA per K-step.
// Double-buffered LDS: loads for tile s+1 issued before tile s compute.
#define BM 64
#define BN 128
#define BK 32
#define LDS_K 48   // bf16 row stride: 96B (multiple of 32B -> aligned v16bf)

__global__ __launch_bounds__(256)
void gemm_kernel(const float* __restrict__ A, const __bf16* __restrict__ Wt,
                 const float* __restrict__ bias, const float* __restrict__ Res,
                 float* __restrict__ C, int N, int K, int do_gelu) {
  __shared__ __align__(32) __bf16 sA[2][BM][LDS_K];
  __shared__ __align__(32) __bf16 sB[2][BN][LDS_K];
  int n0 = blockIdx.x * BN, m0 = blockIdx.y * BM;
  int tid = threadIdx.x, lane = tid & 31, w = tid >> 5;
  int hl = lane >> 4, lr = lane & 15;
  int mo = (w >> 1) * 16, nb = (w & 1) * 64;

  // per-thread staging coordinates (2 float4 of A, 2 uint4 of Wt per step)
  int fid1 = tid + 256;
  int ar0 = tid  >> 3, ac0 = (tid  & 7) * 4;
  int ar1 = fid1 >> 3, ac1 = (fid1 & 7) * 4;
  int bn0 = tid  >> 2, bc0 = (tid  & 3) * 8;
  int bn1 = fid1 >> 2, bc1 = (fid1 & 3) * 8;
  const float*  Ar0 = A  + (size_t)(m0 + ar0) * K + ac0;
  const float*  Ar1 = A  + (size_t)(m0 + ar1) * K + ac1;
  const __bf16* Br0 = Wt + (size_t)(n0 + bn0) * K + bc0;
  const __bf16* Br1 = Wt + (size_t)(n0 + bn1) * K + bc1;

  float4 ra0, ra1; uint4 rb0, rb1;       // staging registers
  auto load_tile = [&](int kofs) {
    ra0 = *(const float4*)(Ar0 + kofs);
    ra1 = *(const float4*)(Ar1 + kofs);
    rb0 = *(const uint4*)(Br0 + kofs);
    rb1 = *(const uint4*)(Br1 + kofs);
  };
  auto store_tile = [&](int bb) {
    union { __bf16 h[4]; uint2 u; } p0, p1;
    p0.h[0] = (__bf16)ra0.x; p0.h[1] = (__bf16)ra0.y;
    p0.h[2] = (__bf16)ra0.z; p0.h[3] = (__bf16)ra0.w;
    p1.h[0] = (__bf16)ra1.x; p1.h[1] = (__bf16)ra1.y;
    p1.h[2] = (__bf16)ra1.z; p1.h[3] = (__bf16)ra1.w;
    *(uint2*)&sA[bb][ar0][ac0] = p0.u;
    *(uint2*)&sA[bb][ar1][ac1] = p1.u;
    *(uint4*)&sB[bb][bn0][bc0] = rb0;
    *(uint4*)&sB[bb][bn1][bc1] = rb1;
  };

  F32x8 acc[4];
#pragma unroll
  for (int t = 0; t < 4; ++t) acc[t].v = (v8f){};

  load_tile(0);
  store_tile(0);
  __syncthreads();

  int nsteps = K / BK;
  for (int s = 0; s < nsteps; ++s) {
    int cur = s & 1;
    bool more = (s + 1) < nsteps;
    if (more) load_tile((s + 1) * BK);   // global loads in flight over compute
    if (s + 2 < nsteps)                   // prefetch tile s+2 of weights
      __builtin_prefetch(Br0 + (s + 2) * BK, 0, 1);

    v16bf af = *(const v16bf*)&sA[cur][mo + lr][hl * 16];
#pragma unroll
    for (int t = 0; t < 4; ++t) {
      v16bf bw = *(const v16bf*)&sB[cur][nb + t * 16 + lr][hl * 16];
      acc[t].v = __builtin_amdgcn_wmma_f32_16x16x32_bf16(
          false, af, false, bw, (short)0, acc[t].v, false, false);
    }
    if (more) store_tile(cur ^ 1);        // single loadcnt wait lands here
    __syncthreads();
  }

#pragma unroll
  for (int t = 0; t < 4; ++t) {
    int col = n0 + nb + t * 16 + lr;
    float bcol = bias ? bias[col] : 0.0f;
#pragma unroll
    for (int r = 0; r < 8; ++r) {
      int row = m0 + mo + r + hl * 8;   // C layout: VGPR r -> row r (+8 hi half)
      float v = acc[t].e[r] + bcol;
      if (do_gelu) v = 0.5f * v * (1.0f + erff(v * 0.70710678118f));
      if (Res) v += Res[(size_t)row * N + col];
      C[(size_t)row * N + col] = v;
    }
  }
}

// --------------------------- flash attention --------------------------------
// One wave per (b, head, 16-row Q tile); kv tiles of 32 with online softmax.
__global__ __launch_bounds__(32)
void attn_kernel(const float* __restrict__ qkv, float* __restrict__ o) {
  __shared__ __align__(32) __bf16 sVt[64][32];  // V tile transposed [hd][kv]
  __shared__ __align__(32) __bf16 sP[16][32];   // probabilities [q][kv]
  int lane = threadIdx.x, hl = lane >> 4, lr = lane & 15;
  int q0 = blockIdx.x * 16, hh = blockIdx.y, b = blockIdx.z;
  size_t mbase = (size_t)b * L_;
  const int qc = hh * HD_, kc = D_ + hh * HD_, vc = 2 * D_ + hh * HD_;
  const float scale = 0.125f;                   // 1/sqrt(HD)

  // Q fragments (scale folded in): K-dim 64 -> two bf16 A-frags
  BF16x16 qa0, qa1;
  {
    const float4* qr =
        (const float4*)(qkv + (mbase + q0 + lr) * S3_ + qc + hl * 16);
#pragma unroll
    for (int j = 0; j < 4; ++j) {
      float4 f0 = qr[j], f1 = qr[8 + j];
      qa0.e[4*j+0] = (__bf16)(f0.x*scale); qa0.e[4*j+1] = (__bf16)(f0.y*scale);
      qa0.e[4*j+2] = (__bf16)(f0.z*scale); qa0.e[4*j+3] = (__bf16)(f0.w*scale);
      qa1.e[4*j+0] = (__bf16)(f1.x*scale); qa1.e[4*j+1] = (__bf16)(f1.y*scale);
      qa1.e[4*j+2] = (__bf16)(f1.z*scale); qa1.e[4*j+3] = (__bf16)(f1.w*scale);
    }
  }

  float mrow[8], lsum[8];
  F32x8 oacc[4];
#pragma unroll
  for (int r = 0; r < 8; ++r) { mrow[r] = -3.0e38f; lsum[r] = 0.f; }
#pragma unroll
  for (int g = 0; g < 4; ++g) oacc[g].v = (v8f){};

  for (int j0 = 0; j0 < q0 + 16; j0 += 32) {
    __syncthreads();
    // stage V^T: sVt[hd][kv] (float4 reads coalesced along hd)
#pragma unroll
    for (int it = 0; it < 16; ++it) {
      int idx = lane + it * 32;                 // 0..511 float4 groups
      int kk = idx >> 4, n4 = (idx & 15) * 4;
      float4 vv = *(const float4*)&qkv[(mbase + j0 + kk) * S3_ + vc + n4];
      sVt[n4 + 0][kk] = (__bf16)vv.x; sVt[n4 + 1][kk] = (__bf16)vv.y;
      sVt[n4 + 2][kk] = (__bf16)vv.z; sVt[n4 + 3][kk] = (__bf16)vv.w;
    }
    // S = Q K^T for two 16-col kv sub-tiles (K rows contiguous -> B frags)
    F32x8 s0, s1; s0.v = (v8f){}; s1.v = (v8f){};
    {
      BF16x16 kb0, kb1;
      const float4* kr =
          (const float4*)(qkv + (mbase + j0 + lr) * S3_ + kc + hl * 16);
#pragma unroll
      for (int j = 0; j < 4; ++j) {
        float4 f0 = kr[j], f1 = kr[8 + j];
        kb0.e[4*j+0]=(__bf16)f0.x; kb0.e[4*j+1]=(__bf16)f0.y;
        kb0.e[4*j+2]=(__bf16)f0.z; kb0.e[4*j+3]=(__bf16)f0.w;
        kb1.e[4*j+0]=(__bf16)f1.x; kb1.e[4*j+1]=(__bf16)f1.y;
        kb1.e[4*j+2]=(__bf16)f1.z; kb1.e[4*j+3]=(__bf16)f1.w;
      }
      s0.v = __builtin_amdgcn_wmma_f32_16x16x32_bf16(false, qa0.v, false, kb0.v,
                                                     (short)0, s0.v, false, false);
      s0.v = __builtin_amdgcn_wmma_f32_16x16x32_bf16(false, qa1.v, false, kb1.v,
                                                     (short)0, s0.v, false, false);
      const float4* kr2 =
          (const float4*)(qkv + (mbase + j0 + 16 + lr) * S3_ + kc + hl * 16);
#pragma unroll
      for (int j = 0; j < 4; ++j) {
        float4 f0 = kr2[j], f1 = kr2[8 + j];
        kb0.e[4*j+0]=(__bf16)f0.x; kb0.e[4*j+1]=(__bf16)f0.y;
        kb0.e[4*j+2]=(__bf16)f0.z; kb0.e[4*j+3]=(__bf16)f0.w;
        kb1.e[4*j+0]=(__bf16)f1.x; kb1.e[4*j+1]=(__bf16)f1.y;
        kb1.e[4*j+2]=(__bf16)f1.z; kb1.e[4*j+3]=(__bf16)f1.w;
      }
      s1.v = __builtin_amdgcn_wmma_f32_16x16x32_bf16(false, qa0.v, false, kb0.v,
                                                     (short)0, s1.v, false, false);
      s1.v = __builtin_amdgcn_wmma_f32_16x16x32_bf16(false, qa1.v, false, kb1.v,
                                                     (short)0, s1.v, false, false);
    }
    // causal mask + online softmax (row stats reduced across 16-lane halves)
    int c0 = j0 + lr, c1 = j0 + 16 + lr;
#pragma unroll
    for (int r = 0; r < 8; ++r) {
      int rowg = q0 + r + hl * 8;
      float a0 = (c0 <= rowg) ? s0.e[r] : -3.0e38f;
      float a1 = (c1 <= rowg) ? s1.e[r] : -3.0e38f;
      float mt = fmaxf(a0, a1);
      for (int m = 8; m >= 1; m >>= 1) mt = fmaxf(mt, __shfl_xor(mt, m, 32));
      float nm = fmaxf(mrow[r], mt);
      float fac = __expf(mrow[r] - nm);
      float p0 = __expf(a0 - nm), p1 = __expf(a1 - nm);
      float ps = p0 + p1;
      for (int m = 8; m >= 1; m >>= 1) ps += __shfl_xor(ps, m, 32);
      lsum[r] = lsum[r] * fac + ps;
      mrow[r] = nm;
#pragma unroll
      for (int g = 0; g < 4; ++g) oacc[g].e[r] *= fac;
      sP[r + hl * 8][lr]      = (__bf16)p0;
      sP[r + hl * 8][16 + lr] = (__bf16)p1;
    }
    __syncthreads();
    // O += P @ V : P re-laid-out via LDS into an A-frag; V^T gives B-frags
    v16bf pf = *(const v16bf*)&sP[lr][hl * 16];
#pragma unroll
    for (int g = 0; g < 4; ++g) {
      v16bf vb = *(const v16bf*)&sVt[g * 16 + lr][hl * 16];
      oacc[g].v = __builtin_amdgcn_wmma_f32_16x16x32_bf16(
          false, pf, false, vb, (short)0, oacc[g].v, false, false);
    }
  }

#pragma unroll
  for (int g = 0; g < 4; ++g)
#pragma unroll
    for (int r = 0; r < 8; ++r) {
      int rowg = q0 + r + hl * 8;
      o[(mbase + rowg) * D_ + hh * HD_ + g * 16 + lr] = oacc[g].e[r] / lsum[r];
    }
}

// ----------------------------- orchestration --------------------------------
extern "C" void kernel_launch(void* const* d_in, const int* in_sizes, int n_in,
                              void* d_out, int out_size, void* d_ws, size_t ws_size,
                              hipStream_t stream) {
  (void)in_sizes; (void)n_in; (void)out_size; (void)ws_size;
  const int*   x       = (const int*)  d_in[0];
  const float* tok_emb = (const float*)d_in[1];
  const float* pos_emb = (const float*)d_in[2];
  const float* ln1_g   = (const float*)d_in[3];
  const float* ln1_b   = (const float*)d_in[4];
  const float* qkv_w   = (const float*)d_in[5];
  const float* qkv_b   = (const float*)d_in[6];
  const float* out_w   = (const float*)d_in[7];
  const float* out_b   = (const float*)d_in[8];
  const float* ln2_g   = (const float*)d_in[9];
  const float* ln2_b   = (const float*)d_in[10];
  const float* mlp_w1  = (const float*)d_in[11];
  const float* mlp_b1  = (const float*)d_in[12];
  const float* mlp_w2  = (const float*)d_in[13];
  const float* mlp_b2  = (const float*)d_in[14];
  const float* lnf_g   = (const float*)d_in[15];
  const float* lnf_b   = (const float*)d_in[16];
  const float* head_w  = (const float*)d_in[17];

  float* h   = (float*)d_ws;                 // [M, D]
  float* a   = h   + (size_t)M_ * D_;        // [M, D]
  float* qkv = a   + (size_t)M_ * D_;        // [M, 3D]
  float* o   = qkv + (size_t)M_ * S3_;       // [M, D]
  float* m1  = o   + (size_t)M_ * D_;        // [M, DM]
  __bf16* wt = (__bf16*)(m1 + (size_t)M_ * DM_);  // [N_max=32000, K] bf16

  embed_kernel<<<M_, 128, 0, stream>>>(x, tok_emb, pos_emb, h);
  for (int i = 0; i < NL_; ++i) {
    ln_kernel<<<M_, 128, 0, stream>>>(h, ln1_g + i * D_, ln1_b + i * D_, a);

    wconv_kernel<<<dim3(D_ / 64, S3_ / 64), 256, 0, stream>>>(
        qkv_w + (size_t)i * D_ * S3_, wt, D_, S3_);
    gemm_kernel<<<dim3(S3_ / BN, M_ / BM), 256, 0, stream>>>(
        a, wt, qkv_b + (size_t)i * S3_, nullptr, qkv, S3_, D_, 0);

    attn_kernel<<<dim3(L_ / 16, NH_, B_), 32, 0, stream>>>(qkv, o);

    wconv_kernel<<<dim3(D_ / 64, D_ / 64), 256, 0, stream>>>(
        out_w + (size_t)i * D_ * D_, wt, D_, D_);
    gemm_kernel<<<dim3(D_ / BN, M_ / BM), 256, 0, stream>>>(
        o, wt, out_b + (size_t)i * D_, h, h, D_, D_, 0);

    ln_kernel<<<M_, 128, 0, stream>>>(h, ln2_g + i * D_, ln2_b + i * D_, a);

    wconv_kernel<<<dim3(D_ / 64, DM_ / 64), 256, 0, stream>>>(
        mlp_w1 + (size_t)i * D_ * DM_, wt, D_, DM_);
    gemm_kernel<<<dim3(DM_ / BN, M_ / BM), 256, 0, stream>>>(
        a, wt, mlp_b1 + (size_t)i * DM_, nullptr, m1, DM_, D_, 1);

    wconv_kernel<<<dim3(DM_ / 64, D_ / 64), 256, 0, stream>>>(
        mlp_w2 + (size_t)i * DM_ * D_, wt, DM_, D_);
    gemm_kernel<<<dim3(D_ / BN, M_ / BM), 256, 0, stream>>>(
        m1, wt, mlp_b2 + (size_t)i * D_, h, h, D_, DM_, 0);
  }
  ln_kernel<<<M_, 128, 0, stream>>>(h, lnf_g, lnf_b, a);
  wconv_kernel<<<dim3(D_ / 64, V_ / 64), 256, 0, stream>>>(head_w, wt, D_, V_);
  gemm_kernel<<<dim3(V_ / BN, M_ / BM), 256, 0, stream>>>(
      a, wt, nullptr, nullptr, (float*)d_out, V_, D_, 0);
}
